// Decoder_6536940224687
// MI455X (gfx1250) — matrix operand order repaired and checked by reading, not compiled, not run on hardware
//
#include <hip/hip_runtime.h>
#include <math.h>

typedef __attribute__((ext_vector_type(16))) __bf16 v16bf;
typedef __attribute__((ext_vector_type(8)))  __bf16 v8bf;
typedef __attribute__((ext_vector_type(8)))  float  v8f;

#define N_MELS 80
#define RF 2
#define ENC 512
#define DECH 1024
#define ATT 128
#define PRE 256
#define LOC_F 32
#define LOC_K 31
#define BB 32
#define TT 512
#define TMEL 800
#define NSTEPS 400
#define K1 1792   // prenet(256) + ctx(512) + h1(1024)
#define K2 2048   // h1n(1024) + h2(1024)
#define KOUT 1536 // h2n(1024) + ctx(512)

// ---------------------------------------------------------------------------
// WMMA helper: D(32x16) += A(32xK) * W(16xK)^T using v_wmma_f32_16x16x32_bf16
// A row-major (32 x lda), W row-major (rows = output cols, ld = ldb).
// Fragment layouts per CDNA5 ISA 7.12.2 (16-bit A 16x32; B 32x16; 32-bit C/D).
// ---------------------------------------------------------------------------
__device__ __forceinline__ void gemm2m(const __bf16* __restrict__ A, int lda,
                                       const __bf16* __restrict__ Bw, int ldb,
                                       int K, v8f& acc0, v8f& acc1) {
  const int lane = threadIdx.x & 31;
  const int r = lane & 15;        // A row within tile / B output column
  const int g = lane >> 4;        // lane-half select
  const __bf16* pa0 = A + r * lda + 8 * g;
  const __bf16* pa1 = A + (16 + r) * lda + 8 * g;
  const __bf16* pb  = Bw + r * ldb + 8 * g;
  for (int kk = 0; kk < K; kk += 32) {
    union { v16bf v; v8bf h[2]; } a0, a1, b;
    a0.h[0] = *(const v8bf*)(pa0 + kk);
    a0.h[1] = *(const v8bf*)(pa0 + kk + 16);
    a1.h[0] = *(const v8bf*)(pa1 + kk);
    a1.h[1] = *(const v8bf*)(pa1 + kk + 16);
    b.h[0]  = *(const v8bf*)(pb  + kk);
    b.h[1]  = *(const v8bf*)(pb  + kk + 16);
    acc0 = __builtin_amdgcn_wmma_f32_16x16x32_bf16(false, a0.v, false, b.v,
                                                   (short)0, acc0, false, false);
    acc1 = __builtin_amdgcn_wmma_f32_16x16x32_bf16(false, a1.v, false, b.v,
                                                   (short)0, acc1, false, false);
  }
}

// ---------------------------------------------------------------------------
// One-time setup kernels
// ---------------------------------------------------------------------------
__global__ void k_cvt_bf16(const float* __restrict__ s, __bf16* __restrict__ d, int n) {
  for (int i = blockIdx.x * blockDim.x + threadIdx.x; i < n; i += gridDim.x * blockDim.x)
    d[i] = (__bf16)s[i];
}

__global__ void k_fill_f32(float* p, float v, int n) {
  for (int i = blockIdx.x * blockDim.x + threadIdx.x; i < n; i += gridDim.x * blockDim.x)
    p[i] = v;
}

__global__ void k_fill_bf16(__bf16* p, int n) {
  for (int i = blockIdx.x * blockDim.x + threadIdx.x; i < n; i += gridDim.x * blockDim.x)
    p[i] = (__bf16)0.0f;
}

__global__ void k_init_aw(float* aw, float* acum) {
  for (int i = blockIdx.x * blockDim.x + threadIdx.x; i < BB * TT; i += gridDim.x * blockDim.x) {
    aw[i]   = ((i & (TT - 1)) == 0) ? 1.0f : 0.0f;
    acum[i] = 0.0f;
  }
}

// dst (4096 x (kih+khh)) = [w_ih | w_hh] in bf16
__global__ void k_build_wcat(const float* __restrict__ wih, const float* __restrict__ whh,
                             __bf16* __restrict__ dst, int kih, int khh, int n) {
  int K = kih + khh;
  for (int i = blockIdx.x * blockDim.x + threadIdx.x; i < n; i += gridDim.x * blockDim.x) {
    int row = i / K, k = i - row * K;
    float v = (k < kih) ? wih[row * kih + k] : whh[row * khh + (k - kih)];
    dst[i] = (__bf16)v;
  }
}

// Wout (176 x 1536): rows 0..159 = mel_w, row 160 = stop_w, rest zero
__global__ void k_build_wout(const float* __restrict__ melw, const float* __restrict__ stopw,
                             __bf16* __restrict__ dst) {
  int n = 176 * KOUT;
  for (int i = blockIdx.x * blockDim.x + threadIdx.x; i < n; i += gridDim.x * blockDim.x) {
    int row = i / KOUT, k = i - row * KOUT;
    float v = 0.0f;
    if (row < 160) v = melw[row * KOUT + k];
    else if (row == 160) v = stopw[k];
    dst[i] = (__bf16)v;
  }
}

// pm (B*T, ATT) = memory_bf (B*T, 512) @ wm_bf(128,512)^T   [one-time WMMA GEMM]
__global__ __launch_bounds__(256) void k_pm(const __bf16* __restrict__ memb,
                                            const __bf16* __restrict__ wmb,
                                            float* __restrict__ pm) {
  int w = blockIdx.x * 8 + (threadIdx.x >> 5);  // 0..4095
  int mt = w >> 3;                              // 32-row block, 0..511
  int nt = w & 7;                               // 16-col block, 0..7
  v8f a0 = {}, a1 = {};
  gemm2m(memb + mt * 32 * ENC, ENC, wmb + nt * 16 * ENC, ENC, ENC, a0, a1);
  int lane = threadIdx.x & 31, g = lane >> 4, col = nt * 16 + (lane & 15);
  for (int vr = 0; vr < 8; vr++) {
    pm[(mt * 32 + vr + 8 * g) * ATT + col]      = a0[vr];
    pm[(mt * 32 + 16 + vr + 8 * g) * ATT + col] = a1[vr];
  }
}

// ---------------------------------------------------------------------------
// Per-step kernels
// ---------------------------------------------------------------------------
// Prenet: two fused 32x256 WMMA GEMMs (K=160 then 256) in a single workgroup.
__global__ __launch_bounds__(256) void k_prenet(const float* __restrict__ mel_t,
                                                const __bf16* __restrict__ pw1b,
                                                const float* __restrict__ pb1,
                                                const __bf16* __restrict__ pw2b,
                                                const float* __restrict__ pb2,
                                                __bf16* __restrict__ xcat, int s) {
  __shared__ __bf16 Ain[BB * 160];
  __shared__ __bf16 X1[BB * PRE];
  int tid = threadIdx.x;
  for (int i = tid; i < BB * 160; i += 256) {
    float v = 0.0f;
    if (s > 0) {
      int b = i / 160, k = i - b * 160;
      v = mel_t[b * (TMEL * N_MELS) + (s - 1) * 160 + k];
    }
    Ain[i] = (__bf16)v;
  }
  __syncthreads();
  int wv = tid >> 5, lane = tid & 31, g = lane >> 4, nn = lane & 15;
  for (int nt = wv; nt < 16; nt += 8) {
    v8f a0 = {}, a1 = {};
    gemm2m(Ain, 160, pw1b + nt * 16 * 160, 160, 160, a0, a1);
    int col = nt * 16 + nn;
    float bias = pb1[col];
    for (int vr = 0; vr < 8; vr++) {
      int m = vr + 8 * g;
      X1[m * PRE + col]        = (__bf16)fmaxf(a0[vr] + bias, 0.0f);
      X1[(16 + m) * PRE + col] = (__bf16)fmaxf(a1[vr] + bias, 0.0f);
    }
  }
  __syncthreads();
  for (int nt = wv; nt < 16; nt += 8) {
    v8f a0 = {}, a1 = {};
    gemm2m(X1, PRE, pw2b + nt * 16 * PRE, PRE, PRE, a0, a1);
    int col = nt * 16 + nn;
    float bias = pb2[col];
    for (int vr = 0; vr < 8; vr++) {
      int m = vr + 8 * g;
      xcat[m * K1 + col]        = (__bf16)fmaxf(a0[vr] + bias, 0.0f);
      xcat[(16 + m) * K1 + col] = (__bf16)fmaxf(a1[vr] + bias, 0.0f);
    }
  }
}

// gates (32,4096) = X(32,K) @ Wcat(4096,K)^T + bias  — 256 waves, 32x16 tile/wave
__global__ __launch_bounds__(256) void k_lstm_gemm(const __bf16* __restrict__ X,
                                                   const __bf16* __restrict__ W,
                                                   const float* __restrict__ bias,
                                                   float* __restrict__ gates, int K) {
  int w = blockIdx.x * 8 + (threadIdx.x >> 5);  // 0..255 -> n-tile
  int n0 = w * 16;
  v8f a0 = {}, a1 = {};
  gemm2m(X, K, W + (long)n0 * K, K, K, a0, a1);
  int lane = threadIdx.x & 31, g = lane >> 4, col = n0 + (lane & 15);
  float bv = bias[col];
  for (int vr = 0; vr < 8; vr++) {
    gates[(vr + 8 * g) * 4096 + col]      = a0[vr] + bv;
    gates[(16 + vr + 8 * g) * 4096 + col] = a1[vr] + bv;
  }
}

// LSTM elementwise: writes c in place and new h (bf16) to two destinations.
__global__ void k_lstm_cell(const float* __restrict__ gates, float* __restrict__ c,
                            __bf16* __restrict__ hd1, int ld1, int off1,
                            __bf16* __restrict__ hd2, int ld2, int off2) {
  int i = blockIdx.x * blockDim.x + threadIdx.x;
  if (i >= BB * DECH) return;
  int b = i >> 10, n = i & 1023;
  const float* gr = gates + b * 4096;
  float gi = gr[n], gf = gr[1024 + n], gg = gr[2048 + n], go = gr[3072 + n];
  float si = 1.0f / (1.0f + __expf(-gi));
  float sf = 1.0f / (1.0f + __expf(-gf));
  float so = 1.0f / (1.0f + __expf(-go));
  float cn = sf * c[i] + si * tanhf(gg);
  float hn = so * tanhf(cn);
  c[i] = cn;
  hd1[b * ld1 + off1 + n] = (__bf16)hn;
  hd2[b * ld2 + off2 + n] = (__bf16)hn;
}

// Fused location-sensitive attention (one block per batch element):
// loc-conv + energies + masked softmax + alignment/acum update + context.
__global__ __launch_bounds__(256) void k_attn(const __bf16* __restrict__ xcat2,
                                              const __bf16* __restrict__ wqb,
                                              const float* __restrict__ conv_w,
                                              const float* __restrict__ wl,
                                              const float* __restrict__ vvec,
                                              const float* __restrict__ pm,
                                              const float* __restrict__ memf,
                                              const int* __restrict__ mlen,
                                              float* __restrict__ aw_g,
                                              float* __restrict__ acum_g,
                                              float* __restrict__ align_out,
                                              __bf16* __restrict__ decb,
                                              __bf16* __restrict__ xcat, int s) {
  __shared__ float aw_s[TT + 32], ac_s[TT + 32];
  __shared__ float pq_s[ATT];
  __shared__ float cw_s[LOC_F * 2 * LOC_K];
  __shared__ float wl_s[ATT * LOC_F];
  __shared__ float v_s[ATT];
  __shared__ float e_s[TT];
  __shared__ float red[256];
  int b = blockIdx.x, tid = threadIdx.x;

  for (int i = tid; i < TT + 32; i += 256) {
    int t = i - 16;
    bool in = (t >= 0 && t < TT);
    aw_s[i] = in ? aw_g[b * TT + t] : 0.0f;
    ac_s[i] = in ? acum_g[b * TT + t] : 0.0f;
  }
  for (int i = tid; i < LOC_F * 2 * LOC_K; i += 256) cw_s[i] = conv_w[i];
  for (int i = tid; i < ATT * LOC_F; i += 256) wl_s[i] = wl[i];
  if (tid < ATT) {
    v_s[tid] = vvec[tid];
    const __bf16* h2 = xcat2 + b * K2 + 1024;  // h2_new
    const __bf16* wr = wqb + tid * DECH;
    float acc = 0.0f;
    for (int k = 0; k < DECH; k++) acc += (float)h2[k] * (float)wr[k];
    pq_s[tid] = acc;
  }
  __syncthreads();

  int len = mlen[b];
  for (int t = tid; t < TT; t += 256) {
    float locf[LOC_F];
    for (int f = 0; f < LOC_F; f++) {
      float acc = 0.0f;
      const float* cwf = cw_s + f * 62;
      for (int k = 0; k < LOC_K; k++) {
        acc += aw_s[t + k + 1] * cwf[k];
        acc += ac_s[t + k + 1] * cwf[LOC_K + k];
      }
      locf[f] = acc;
    }
    float e = 0.0f;
    const float* pmrow = pm + (b * TT + t) * ATT;
    for (int a = 0; a < ATT; a++) {
      float sacc = pq_s[a] + pmrow[a];
      const float* wla = wl_s + a * LOC_F;
      for (int f = 0; f < LOC_F; f++) sacc += locf[f] * wla[f];
      e += v_s[a] * tanhf(sacc);
    }
    e_s[t] = (t < len) ? e : -1e9f;
  }
  __syncthreads();

  red[tid] = fmaxf(e_s[tid], e_s[tid + 256]);
  __syncthreads();
  for (int off = 128; off > 0; off >>= 1) {
    if (tid < off) red[tid] = fmaxf(red[tid], red[tid + off]);
    __syncthreads();
  }
  float maxv = red[0];
  __syncthreads();
  float s0 = __expf(e_s[tid] - maxv), s1 = __expf(e_s[tid + 256] - maxv);
  e_s[tid] = s0; e_s[tid + 256] = s1;
  red[tid] = s0 + s1;
  __syncthreads();
  for (int off = 128; off > 0; off >>= 1) {
    if (tid < off) red[tid] += red[tid + off];
    __syncthreads();
  }
  float inv = 1.0f / red[0];
  __syncthreads();
  for (int t = tid; t < TT; t += 256) {
    float a = e_s[t] * inv;
    e_s[t] = a;
    aw_g[b * TT + t] = a;
    acum_g[b * TT + t] += a;
    align_out[((long)b * NSTEPS + s) * TT + t] = a;
  }
  __syncthreads();
  for (int e0 = tid; e0 < ENC; e0 += 256) {
    float acc = 0.0f;
    const float* mrow = memf + (long)b * TT * ENC + e0;
    for (int t = 0; t < TT; t++) acc += e_s[t] * mrow[t * ENC];
    decb[b * KOUT + 1024 + e0] = (__bf16)acc;   // ctx for output proj
    xcat[b * K1 + 256 + e0]    = (__bf16)acc;   // ctx for next LSTM1 input
  }
}

// Output projection: (32 x 176) = dec(32,1536) @ Wout(176,1536)^T, scatter mel/stop.
__global__ __launch_bounds__(256) void k_out(const __bf16* __restrict__ decb,
                                             const __bf16* __restrict__ woutb,
                                             const float* __restrict__ melb,
                                             const float* __restrict__ stopb,
                                             float* __restrict__ out_mel,
                                             float* __restrict__ out_stop, int s) {
  int wv = threadIdx.x >> 5, lane = threadIdx.x & 31, g = lane >> 4, nn = lane & 15;
  for (int nt = wv; nt < 11; nt += 8) {
    v8f a0 = {}, a1 = {};
    gemm2m(decb, KOUT, woutb + nt * 16 * KOUT, KOUT, KOUT, a0, a1);
    int col = nt * 16 + nn;
    for (int vr = 0; vr < 8; vr++) {
      int b0 = vr + 8 * g, b1 = 16 + b0;
      if (col < 160) {
        float bias = melb[col];
        out_mel[b0 * (TMEL * N_MELS) + s * 160 + col] = a0[vr] + bias;
        out_mel[b1 * (TMEL * N_MELS) + s * 160 + col] = a1[vr] + bias;
      } else if (col == 160) {
        float bias = stopb[0];
        out_stop[b0 * NSTEPS + s] = a0[vr] + bias;
        out_stop[b1 * NSTEPS + s] = a1[vr] + bias;
      }
    }
  }
}

// ---------------------------------------------------------------------------
static inline int nblk(long n) { long b = (n + 255) / 256; return b > 2048 ? 2048 : (int)b; }

extern "C" void kernel_launch(void* const* d_in, const int* in_sizes, int n_in,
                              void* d_out, int out_size, void* d_ws, size_t ws_size,
                              hipStream_t stream) {
  const float* memory  = (const float*)d_in[0];
  const float* mel_t   = (const float*)d_in[1];
  const int*   mlen    = (const int*)d_in[2];
  const float* pw1     = (const float*)d_in[3];
  const float* pb1     = (const float*)d_in[4];
  const float* pw2     = (const float*)d_in[5];
  const float* pb2     = (const float*)d_in[6];
  const float* w_ih1   = (const float*)d_in[7];
  const float* w_hh1   = (const float*)d_in[8];
  const float* b1      = (const float*)d_in[9];
  const float* w_ih2   = (const float*)d_in[10];
  const float* w_hh2   = (const float*)d_in[11];
  const float* b2      = (const float*)d_in[12];
  const float* wq      = (const float*)d_in[13];
  const float* wm      = (const float*)d_in[14];
  const float* conv_w  = (const float*)d_in[15];
  const float* wl      = (const float*)d_in[16];
  const float* vv      = (const float*)d_in[17];
  const float* mel_w   = (const float*)d_in[18];
  const float* mel_b   = (const float*)d_in[19];
  const float* stop_w  = (const float*)d_in[20];
  const float* stop_b  = (const float*)d_in[21];

  float* out_mel  = (float*)d_out;                       // (32,800,80)
  float* out_stop = out_mel + (long)BB * TMEL * N_MELS;  // (32,400,1)
  float* out_alg  = out_stop + (long)BB * NSTEPS;        // (32,400,512)

  // Workspace bump allocator (256B aligned segments)
  char* p = (char*)d_ws;
  auto take = [&](size_t bytes) -> void* {
    void* r = (void*)p;
    p += (bytes + 255) & ~(size_t)255;
    return r;
  };
  float*  pm    = (float*)take((size_t)BB * TT * ATT * 4);
  float*  gates = (float*)take((size_t)BB * 4096 * 4);
  float*  c1    = (float*)take((size_t)BB * DECH * 4);
  float*  c2    = (float*)take((size_t)BB * DECH * 4);
  float*  aw    = (float*)take((size_t)BB * TT * 4);
  float*  acum  = (float*)take((size_t)BB * TT * 4);
  __bf16* memb  = (__bf16*)take((size_t)BB * TT * ENC * 2);
  __bf16* wcat1 = (__bf16*)take((size_t)4096 * K1 * 2);
  __bf16* wcat2 = (__bf16*)take((size_t)4096 * K2 * 2);
  __bf16* woutb = (__bf16*)take((size_t)176 * KOUT * 2);
  __bf16* pw1b  = (__bf16*)take((size_t)PRE * 160 * 2);
  __bf16* pw2b  = (__bf16*)take((size_t)PRE * PRE * 2);
  __bf16* wqb   = (__bf16*)take((size_t)ATT * DECH * 2);
  __bf16* wmb   = (__bf16*)take((size_t)ATT * ENC * 2);
  __bf16* xcat  = (__bf16*)take((size_t)BB * K1 * 2);
  __bf16* xcat2 = (__bf16*)take((size_t)BB * K2 * 2);
  __bf16* decb  = (__bf16*)take((size_t)BB * KOUT * 2);
  if ((size_t)(p - (char*)d_ws) > ws_size) return;  // insufficient scratch

  // ---- one-time: bf16 conversions / fused weights / pm / state init ----
  int n;
  n = BB * TT * ENC;  k_cvt_bf16<<<nblk(n), 256, 0, stream>>>(memory, memb, n);
  n = PRE * 160;      k_cvt_bf16<<<nblk(n), 256, 0, stream>>>(pw1, pw1b, n);
  n = PRE * PRE;      k_cvt_bf16<<<nblk(n), 256, 0, stream>>>(pw2, pw2b, n);
  n = ATT * DECH;     k_cvt_bf16<<<nblk(n), 256, 0, stream>>>(wq, wqb, n);
  n = ATT * ENC;      k_cvt_bf16<<<nblk(n), 256, 0, stream>>>(wm, wmb, n);
  n = 4096 * K1;      k_build_wcat<<<nblk(n), 256, 0, stream>>>(w_ih1, w_hh1, wcat1, 768, 1024, n);
  n = 4096 * K2;      k_build_wcat<<<nblk(n), 256, 0, stream>>>(w_ih2, w_hh2, wcat2, 1024, 1024, n);
  k_build_wout<<<nblk(176 * KOUT), 256, 0, stream>>>(mel_w, stop_w, woutb);
  n = BB * DECH;      k_fill_f32<<<nblk(n), 256, 0, stream>>>(c1, 0.0f, n);
                      k_fill_f32<<<nblk(n), 256, 0, stream>>>(c2, 0.0f, n);
  k_init_aw<<<nblk(BB * TT), 256, 0, stream>>>(aw, acum);
  n = BB * K1;        k_fill_bf16<<<nblk(n), 256, 0, stream>>>(xcat, n);
  n = BB * K2;        k_fill_bf16<<<nblk(n), 256, 0, stream>>>(xcat2, n);
  n = BB * KOUT;      k_fill_bf16<<<nblk(n), 256, 0, stream>>>(decb, n);
  k_pm<<<512, 256, 0, stream>>>(memb, wmb, pm);  // pm = memory @ wm^T (WMMA)

  // ---- sequential decode: 400 teacher-forced steps ----
  for (int s = 0; s < NSTEPS; s++) {
    k_prenet<<<1, 256, 0, stream>>>(mel_t, pw1b, pb1, pw2b, pb2, xcat, s);
    k_lstm_gemm<<<32, 256, 0, stream>>>(xcat, wcat1, b1, gates, K1);
    k_lstm_cell<<<128, 256, 0, stream>>>(gates, c1, xcat, K1, 768, xcat2, K2, 0);
    k_lstm_gemm<<<32, 256, 0, stream>>>(xcat2, wcat2, b2, gates, K2);
    k_lstm_cell<<<128, 256, 0, stream>>>(gates, c2, xcat2, K2, 1024, decb, KOUT, 0);
    k_attn<<<32, 256, 0, stream>>>(xcat2, wqb, conv_w, wl, vv, pm, memory, mlen,
                                   aw, acum, out_alg, decb, xcat, s);
    k_out<<<1, 256, 0, stream>>>(decb, woutb, mel_b, stop_b, out_mel, out_stop, s);
  }
}
